// mLSTMLayer_33990371180673
// MI455X (gfx1250) — compile-verified, matrix operand order checked by reading, not run on hardware
//
#include <hip/hip_runtime.h>
#include <hip/hip_bf16.h>

// ---- dims ----
#define B_  64
#define F_  1024
#define H_  4
#define DH_ 512
#define KCONV_ 4
#define U_  2048      // 2*F
#define UP_ 4096      // 2*U

typedef float  v8f   __attribute__((ext_vector_type(8)));
typedef __bf16 v16bf __attribute__((ext_vector_type(16)));
typedef int    v4i   __attribute__((ext_vector_type(4)));
typedef __bf16 bf16;

#if defined(__has_builtin)
#if __has_builtin(__builtin_amdgcn_global_load_async_to_lds_b128) && \
    __has_builtin(__builtin_amdgcn_s_wait_asynccnt)
#define HAVE_ASYNC_LDS 1
#endif
#endif

// ---------------------------------------------------------------------------
// Weight pre-pack: in f32 row-major (K x N)  ->  out bf16 (N x K) [N-major]
// Tiled 32x32 via LDS so both global sides are coalesced.
// ---------------------------------------------------------------------------
__global__ __launch_bounds__(256) void transpose_convert(
    const float* __restrict__ in, bf16* __restrict__ out, int K, int N)
{
  __shared__ float tile[32][33];
  const int kt = blockIdx.y * 32, nt = blockIdx.x * 32;
  const int tx = threadIdx.x & 31, ty = threadIdx.x >> 5;   // 32 x 8
#pragma unroll
  for (int i = 0; i < 32; i += 8)
    tile[ty + i][tx] = in[(size_t)(kt + ty + i) * N + nt + tx];
  __syncthreads();
#pragma unroll
  for (int i = 0; i < 32; i += 8)
    out[(size_t)(nt + ty + i) * K + kt + tx] = (bf16)tile[tx][ty + i];
}

// Elementwise f32 -> bf16 (for the batch activations)
__global__ __launch_bounds__(256) void convert_bf16(
    const float* __restrict__ in, bf16* __restrict__ out, int n)
{
  int i = blockIdx.x * 256 + threadIdx.x;
  if (i < n) out[i] = (bf16)in[i];
}

// ---------------------------------------------------------------------------
// WMMA GEMM: Out(M,N) f32 = A(M,K) bf16 * Bt(N,K) bf16 [+ bias].
// One wave per 16x16 tile; fragments are two b128 loads per operand,
// sequential in K. v_wmma_f32_16x16x32_bf16, f32 accumulate.
// ---------------------------------------------------------------------------
__global__ __launch_bounds__(256) void gemm_wmma(
    const bf16* __restrict__ A, int lda,
    const bf16* __restrict__ Bt, int ldbt,
    const float* __restrict__ bias,
    float* __restrict__ Out, int ldo,
    int M, int N, int K)
{
  const int lane = threadIdx.x & 31;
  const int wave = threadIdx.x >> 5;
  const int n0 = (blockIdx.x * 8 + wave) * 16;
  const int m0 = blockIdx.y * 16;
  if (n0 >= N || m0 >= M) return;               // wave-uniform: EXEC stays full

  const int half = lane >> 4;                   // 0: lanes 0-15, 1: lanes 16-31
  const int l15  = lane & 15;

  // A frag: lane row = m0+l15, elems 0..7 = K(base..base+7), 8..15 = +16
  const bf16* Ap = A  + (size_t)(m0 + l15) * lda  + half * 8;
  const bf16* Bp = Bt + (size_t)(n0 + l15) * ldbt + half * 8;

  union Frag { v16bf v; uint4 u[2]; };
  v8f c = {};
  for (int k0 = 0; k0 < K; k0 += 32) {
    Frag fa, fb;
    fa.u[0] = *(const uint4*)(Ap + k0);
    fa.u[1] = *(const uint4*)(Ap + k0 + 16);
    fb.u[0] = *(const uint4*)(Bp + k0);
    fb.u[1] = *(const uint4*)(Bp + k0 + 16);
    c = __builtin_amdgcn_wmma_f32_16x16x32_bf16(
        false, fa.v, false, fb.v, (short)0, c, false, false);
  }

  // C/D layout: VGPR r: lanes0-15 -> M=m0+r, lanes16-31 -> M=m0+8+r; N=n0+l15
  const int col  = n0 + l15;
  const int rowb = m0 + half * 8;
  const float bv = bias ? bias[col] : 0.f;
#pragma unroll
  for (int r = 0; r < 8; ++r)
    Out[(size_t)(rowb + r) * ldo + col] = c[r] + bv;
}

// ---------------------------------------------------------------------------
// conv_new = concat(conv_state[:,1:], up_core); conv_act = silu(conv_x)
// Also emits bf16 copies of conv_act and up_core for the q/k/v GEMMs.
// ---------------------------------------------------------------------------
__global__ __launch_bounds__(256) void conv_kernel(
    const float* __restrict__ up, const float* __restrict__ conv_state,
    const float* __restrict__ conv_w, const float* __restrict__ conv_b,
    float* __restrict__ conv_new_out, float* __restrict__ conv_act,
    bf16* __restrict__ b_conv_act, bf16* __restrict__ b_up_core)
{
  int idx = blockIdx.x * blockDim.x + threadIdx.x;     // B*U
  if (idx >= B_ * U_) return;
  int b = idx >> 11;
  int u = idx & (U_ - 1);
  float x3 = up[(size_t)b * UP_ + U_ + u];             // up_core
  float c0 = conv_state[((size_t)b * KCONV_ + 1) * U_ + u];
  float c1 = conv_state[((size_t)b * KCONV_ + 2) * U_ + u];
  float c2 = conv_state[((size_t)b * KCONV_ + 3) * U_ + u];
  conv_new_out[((size_t)b * KCONV_ + 0) * U_ + u] = c0;
  conv_new_out[((size_t)b * KCONV_ + 1) * U_ + u] = c1;
  conv_new_out[((size_t)b * KCONV_ + 2) * U_ + u] = c2;
  conv_new_out[((size_t)b * KCONV_ + 3) * U_ + u] = x3;
  float cx = c0 * conv_w[u] + c1 * conv_w[U_ + u] + c2 * conv_w[2 * U_ + u]
           + x3 * conv_w[3 * U_ + u] + conv_b[u];
  float act = cx / (1.f + expf(-cx));                  // silu
  conv_act[idx]   = act;
  b_conv_act[idx] = (bf16)act;
  b_up_core[idx]  = (bf16)x3;
}

// ---------------------------------------------------------------------------
// Per (b,h): i_t/f_t reductions over qkv (3U), m_new, i_p, f_p, n_new, den.
// ---------------------------------------------------------------------------
__global__ __launch_bounds__(256) void gates_kernel(
    const float* __restrict__ q, const float* __restrict__ k,
    const float* __restrict__ v,
    const float* __restrict__ Wi, const float* __restrict__ bi,
    const float* __restrict__ Wf, const float* __restrict__ bf_,
    const float* __restrict__ m_old, const float* __restrict__ n_old,
    float* __restrict__ m_new_out, float* __restrict__ n_new_out,
    float* __restrict__ ip_ws, float* __restrict__ fp_ws,
    float* __restrict__ den_ws)
{
  __shared__ float redI[256];
  __shared__ float redF[256];
  __shared__ float bc[3];
  const int tid = threadIdx.x;
  const int bh = blockIdx.x;
  const int b = bh >> 2, h = bh & 3;
  const float* qb = q + (size_t)b * U_;
  const float* kb = k + (size_t)b * U_;
  const float* vb = v + (size_t)b * U_;

  float si = 0.f, sf = 0.f;
  for (int u = tid; u < U_; u += 256) {
    float qv = qb[u], kv = kb[u], vv = vb[u];
    si += qv * Wi[(size_t)u * H_ + h]
        + kv * Wi[(size_t)(U_ + u) * H_ + h]
        + vv * Wi[(size_t)(2 * U_ + u) * H_ + h];
    sf += qv * Wf[(size_t)u * H_ + h]
        + kv * Wf[(size_t)(U_ + u) * H_ + h]
        + vv * Wf[(size_t)(2 * U_ + u) * H_ + h];
  }
  redI[tid] = si; redF[tid] = sf;
  __syncthreads();
  for (int s = 128; s > 0; s >>= 1) {
    if (tid < s) { redI[tid] += redI[tid + s]; redF[tid] += redF[tid + s]; }
    __syncthreads();
  }
  if (tid == 0) {
    float it = redI[0] + bi[h];
    float ft = redF[0] + bf_[h];
    float logf_ = (ft > 0.f) ? -log1pf(expf(-ft)) : (ft - log1pf(expf(ft)));
    float mo = m_old[bh];
    float mn = fmaxf(logf_ + mo, it);
    float ipv = expf(it - mn);
    float fpv = expf(logf_ + mo - mn);
    m_new_out[bh] = mn;
    ip_ws[bh] = ipv; fp_ws[bh] = fpv;
    bc[0] = ipv; bc[1] = fpv; bc[2] = mn;
  }
  __syncthreads();
  const float ipv = bc[0], fpv = bc[1], mn = bc[2];
  const float scale = 0.044194173824159216f;   // 1/sqrt(512)
  float qn = 0.f;
  for (int d = tid; d < DH_; d += 256) {
    float kh = kb[h * DH_ + d] * scale;
    float nn = fpv * n_old[(size_t)bh * DH_ + d] + ipv * kh;
    n_new_out[(size_t)bh * DH_ + d] = nn;
    qn += qb[h * DH_ + d] * nn;
  }
  redI[tid] = qn;
  __syncthreads();
  for (int s = 128; s > 0; s >>= 1) {
    if (tid < s) redI[tid] += redI[tid + s];
    __syncthreads();
  }
  if (tid == 0)
    den_ws[bh] = fmaxf(fabsf(redI[0]), expf(-mn)) + 1e-6f;
}

// ---------------------------------------------------------------------------
// Fused state update (HBM roofline: 536 MB read+write of C):
//   c_new[d1,d2] = f_p*C + (i_p/sqrt(DH))*k[d1]*v[d2];  nom += q[d1]*c_new
// q/k rows staged in LDS via async-to-LDS DMA (fallback: plain loads).
// ---------------------------------------------------------------------------
__global__ __launch_bounds__(256) void cupdate_kernel(
    const float* __restrict__ C,
    const float* __restrict__ q, const float* __restrict__ k,
    const float* __restrict__ v,
    const float* __restrict__ ip, const float* __restrict__ fp,
    const float* __restrict__ den,
    float* __restrict__ c_new, float* __restrict__ h_t)
{
  __shared__ float sq[DH_];
  __shared__ float sk[DH_];
  const int tid = threadIdx.x;
  const int chunk = blockIdx.x & 1;
  const int bh = blockIdx.x >> 1;
  const int b = bh >> 2, h = bh & 3;
  const float scale = 0.044194173824159216f;   // 1/sqrt(512)

  const float* qrow = q + (size_t)b * U_ + h * DH_;
  const float* krow = k + (size_t)b * U_ + h * DH_;
#if defined(HAVE_ASYNC_LDS)
  {
    // 256 threads move 2*DH_ floats as per-lane b128 async copies
    const float* src = (tid < 128) ? (qrow + tid * 4) : (krow + (tid - 128) * 4);
    float*       dst = (tid < 128) ? (sq + tid * 4)   : (sk + (tid - 128) * 4);
    __builtin_amdgcn_global_load_async_to_lds_b128(
        (__attribute__((address_space(1))) v4i*)(void*)src,
        (__attribute__((address_space(3))) v4i*)(void*)dst, 0, 0);
    __builtin_amdgcn_s_wait_asynccnt(0);
  }
#else
  for (int d = tid; d < DH_; d += 256) { sq[d] = qrow[d]; sk[d] = krow[d]; }
#endif
  __syncthreads();

  const float fpv = fp[bh], dv = den[bh];
  const float ipk = ip[bh] * scale;            // fold 1/sqrt(DH) into i_p
  const int d2 = chunk * 256 + tid;
  const float vv = v[(size_t)b * U_ + h * DH_ + d2];
  const float* Crow = C + (size_t)bh * DH_ * DH_ + d2;
  float* Orow = c_new + (size_t)bh * DH_ * DH_ + d2;

  float nom = 0.f;
  for (int d1 = 0; d1 < DH_; ++d1) {
    if (d1 + 8 < DH_)
      __builtin_prefetch(Crow + (size_t)(d1 + 8) * DH_, 0, 1);  // global_prefetch
    float cv = fpv * Crow[(size_t)d1 * DH_] + ipk * sk[d1] * vv;
    Orow[(size_t)d1 * DH_] = cv;
    nom += sq[d1] * cv;
  }
  h_t[(size_t)bh * DH_ + d2] = nom / dv;
}

// ---------------------------------------------------------------------------
// Layernorm over DH, * ln_scale, + skip*conv_act, * silu(up_gate) -> h (bf16)
// ---------------------------------------------------------------------------
__global__ __launch_bounds__(256) void ln_gate_kernel(
    const float* __restrict__ h_t, const float* __restrict__ ln_scale,
    const float* __restrict__ skip, const float* __restrict__ conv_act,
    const float* __restrict__ up, bf16* __restrict__ b_h)
{
  __shared__ float red[256];
  __shared__ float st[2];
  const int tid = threadIdx.x;
  const int bh = blockIdx.x;
  const int b = bh >> 2, h = bh & 3;
  const float* ht = h_t + (size_t)bh * DH_;

  float s = 0.f;
  for (int d = tid; d < DH_; d += 256) s += ht[d];
  red[tid] = s; __syncthreads();
  for (int r = 128; r > 0; r >>= 1) { if (tid < r) red[tid] += red[tid + r]; __syncthreads(); }
  if (tid == 0) st[0] = red[0] * (1.f / DH_);
  __syncthreads();
  const float mu = st[0];

  s = 0.f;
  for (int d = tid; d < DH_; d += 256) { float t = ht[d] - mu; s += t * t; }
  red[tid] = s; __syncthreads();
  for (int r = 128; r > 0; r >>= 1) { if (tid < r) red[tid] += red[tid + r]; __syncthreads(); }
  if (tid == 0) st[1] = rsqrtf(red[0] * (1.f / DH_) + 1e-6f);
  __syncthreads();
  const float rstd = st[1];

  for (int d = tid; d < DH_; d += 256) {
    int u = h * DH_ + d;
    float hn = (ht[d] - mu) * rstd * ln_scale[u];
    float val = hn + skip[u] * conv_act[(size_t)b * U_ + u];
    float g = up[(size_t)b * UP_ + u];                 // up_gate
    val *= g / (1.f + expf(-g));                       // * silu(gate)
    b_h[(size_t)b * U_ + u] = (bf16)val;
  }
}

// ---------------------------------------------------------------------------
extern "C" void kernel_launch(void* const* d_in, const int* in_sizes, int n_in,
                              void* d_out, int out_size, void* d_ws, size_t ws_size,
                              hipStream_t stream) {
  const float* inputs     = (const float*)d_in[0];
  const float* C          = (const float*)d_in[1];
  const float* n          = (const float*)d_in[2];
  const float* m          = (const float*)d_in[3];
  const float* conv_state = (const float*)d_in[4];
  const float* W_up       = (const float*)d_in[5];
  const float* conv_w     = (const float*)d_in[6];
  const float* conv_b     = (const float*)d_in[7];
  const float* Wq         = (const float*)d_in[8];
  const float* bq         = (const float*)d_in[9];
  const float* Wk         = (const float*)d_in[10];
  const float* bk         = (const float*)d_in[11];
  const float* Wv         = (const float*)d_in[12];
  const float* bv         = (const float*)d_in[13];
  const float* Wi         = (const float*)d_in[14];
  const float* bi         = (const float*)d_in[15];
  const float* Wf         = (const float*)d_in[16];
  const float* bf_        = (const float*)d_in[17];
  const float* ln_scale   = (const float*)d_in[18];
  const float* skip       = (const float*)d_in[19];
  const float* W_down     = (const float*)d_in[20];

  // Outputs, concatenated flat in return order
  float* out      = (float*)d_out;
  float* y        = out;                    // 64*1024      = 65536
  float* c_new    = out + 65536;            // 64*4*512*512 = 67108864
  float* n_new    = out + 67174400;         // 64*4*512     = 131072
  float* m_new    = out + 67305472;         // 64*4         = 256
  float* conv_new = out + 67305728;         // 64*4*2048    = 524288

  // Workspace: f32 region then bf16 region
  float* ws       = (float*)d_ws;
  float* up       = ws;                     // 64*4096 = 262144
  float* conv_act = ws + 262144;            // 131072
  float* qw       = ws + 393216;            // 131072
  float* kw       = ws + 524288;            // 131072
  float* vw       = ws + 655360;            // 131072
  float* h_t      = ws + 786432;            // 131072
  float* ipw      = ws + 917504;            // 256
  float* fpw      = ws + 917760;            // 256
  float* denw     = ws + 918016;            // 256
  bf16* bbase     = (bf16*)(ws + 918272);   // 16B-aligned
  bf16* bW_up     = bbase;                  // 4096*1024
  bf16* bWq       = bbase + 4194304;        // 2048*2048
  bf16* bWk       = bbase + 8388608;
  bf16* bWv       = bbase + 12582912;
  bf16* bW_down   = bbase + 16777216;       // 1024*2048
  bf16* b_inputs  = bbase + 18874368;       // 64*1024
  bf16* b_cact    = bbase + 18939904;       // 64*2048
  bf16* b_upcore  = bbase + 19070976;       // 64*2048
  bf16* b_h       = bbase + 19202048;       // 64*2048

  // 0) pre-pack weights: f32 (K,N) -> bf16 (N,K); convert input activations
  transpose_convert<<<dim3(UP_ / 32, F_ / 32), 256, 0, stream>>>(W_up, bW_up, F_, UP_);
  transpose_convert<<<dim3(U_ / 32, U_ / 32), 256, 0, stream>>>(Wq, bWq, U_, U_);
  transpose_convert<<<dim3(U_ / 32, U_ / 32), 256, 0, stream>>>(Wk, bWk, U_, U_);
  transpose_convert<<<dim3(U_ / 32, U_ / 32), 256, 0, stream>>>(Wv, bWv, U_, U_);
  transpose_convert<<<dim3(F_ / 32, U_ / 32), 256, 0, stream>>>(W_down, bW_down, U_, F_);
  convert_bf16<<<(B_ * F_) / 256, 256, 0, stream>>>(inputs, b_inputs, B_ * F_);

  // 1) up = inputs @ W_up           (M=64, N=4096, K=1024)
  gemm_wmma<<<dim3(UP_ / 128, B_ / 16), 256, 0, stream>>>(
      b_inputs, F_, bW_up, F_, nullptr, up, UP_, B_, UP_, F_);

  // 2) conv_new / conv_act (+ bf16 copies)
  conv_kernel<<<(B_ * U_) / 256, 256, 0, stream>>>(
      up, conv_state, conv_w, conv_b, conv_new, conv_act, b_cact, b_upcore);

  // 3) q/k/v GEMMs (M=64, N=2048, K=2048)
  gemm_wmma<<<dim3(U_ / 128, B_ / 16), 256, 0, stream>>>(
      b_cact, U_, bWq, U_, bq, qw, U_, B_, U_, U_);
  gemm_wmma<<<dim3(U_ / 128, B_ / 16), 256, 0, stream>>>(
      b_cact, U_, bWk, U_, bk, kw, U_, B_, U_, U_);
  gemm_wmma<<<dim3(U_ / 128, B_ / 16), 256, 0, stream>>>(
      b_upcore, U_, bWv, U_, bv, vw, U_, B_, U_, U_);

  // 4) gates: i/f, m_new, i_p, f_p, n_new, den
  gates_kernel<<<B_ * H_, 256, 0, stream>>>(
      qw, kw, vw, Wi, bi, Wf, bf_, m, n, m_new, n_new, ipw, fpw, denw);

  // 5) fused c_new + nom -> h_t   (read C once, write c_new once)
  cupdate_kernel<<<B_ * H_ * 2, 256, 0, stream>>>(
      C, qw, kw, vw, ipw, fpw, denw, c_new, h_t);

  // 6) layernorm + skip + silu gating -> h (bf16)
  ln_gate_kernel<<<B_ * H_, 256, 0, stream>>>(
      h_t, ln_scale, skip, conv_act, up, b_h);

  // 7) y = h @ W_down  (M=64, N=1024, K=2048)
  gemm_wmma<<<dim3(F_ / 128, B_ / 16), 256, 0, stream>>>(
      b_h, U_, bW_down, F_, nullptr, y, F_, B_, F_, U_);
}
